// VirtualNode_24927990186019
// MI455X (gfx1250) — matrix operand order repaired and checked by reading, not compiled, not run on hardware
//
#include <hip/hip_runtime.h>

typedef __attribute__((ext_vector_type(16))) _Float16 v16h;
typedef __attribute__((ext_vector_type(8)))  float    v8f;

#define BGRAPHS 1024
#define HDIM 64
#define NODES_PER_BLOCK 1024
#define EPSF 1e-8f

// ---------------------------------------------------------------- zero ws
__global__ __launch_bounds__(256) void vn_zero_kernel(float* __restrict__ p, int n) {
    int i = blockIdx.x * 256 + threadIdx.x;
    if (i < n) p[i] = 0.0f;
}

// ------------------------------------------------- per-graph centroid sums
__global__ __launch_bounds__(256) void vn_center_partial(
    const float* __restrict__ pos, const int* __restrict__ batch2,
    float* __restrict__ center_sum, int n) {
    __shared__ float sC[BGRAPHS * 4];
    int tid = threadIdx.x;
    for (int i = tid; i < BGRAPHS * 4; i += 256) sC[i] = 0.0f;
    __syncthreads();
    int start = blockIdx.x * NODES_PER_BLOCK;
    for (int i = 0; i < NODES_PER_BLOCK / 256; ++i) {
        int node = start + i * 256 + tid;
        if (node < n) {
            int g = batch2[2 * node];        // low dword of sorted int64 batch id
            atomicAdd(&sC[g * 4 + 0], pos[node * 3 + 0]);
            atomicAdd(&sC[g * 4 + 1], pos[node * 3 + 1]);
            atomicAdd(&sC[g * 4 + 2], pos[node * 3 + 2]);
            atomicAdd(&sC[g * 4 + 3], 1.0f);
        }
    }
    __syncthreads();
    int last = min(start + NODES_PER_BLOCK, n) - 1;
    int gmin = batch2[2 * start];
    int gmax = batch2[2 * last];
    int span = (gmax - gmin + 1) * 4;        // batch sorted -> tiny span
    for (int i = tid; i < span; i += 256) {
        float v = sC[gmin * 4 + i];
        if (v != 0.0f) atomicAdd(&center_sum[gmin * 4 + i], v);
    }
}

__global__ __launch_bounds__(256) void vn_center_final(
    const float* __restrict__ cs, float* __restrict__ center) {
    int b = blockIdx.x * 256 + threadIdx.x;
    if (b < BGRAPHS) {
        float cnt = fmaxf(cs[b * 4 + 3], 1.0f);
        center[b * 3 + 0] = cs[b * 4 + 0] / cnt;
        center[b * 3 + 1] = cs[b * 4 + 1] / cnt;
        center[b * 3 + 2] = cs[b * 4 + 2] / cnt;
    }
}

// -------------------------------------------------------- main fused WMMA
// 8 waves/block, each wave owns 8 tiles of 16 nodes -> 1024 nodes/block.
__global__ __launch_bounds__(256) void vn_main(
    const float* __restrict__ feat, const float* __restrict__ pos,
    const int* __restrict__ batch2,
    const float* __restrict__ W1, const float* __restrict__ b1v,
    const float* __restrict__ W2, const float* __restrict__ b2v,
    const float* __restrict__ center, float* __restrict__ vn_sum, int n) {

    __shared__ _Float16 sW1[HDIM * HDIM];     // B operand: [k][n]
    __shared__ _Float16 sW2[HDIM * 16];       // [k][n], n>=4 zero padded
    __shared__ float    sB1[HDIM];
    __shared__ float    sB2[4];
    __shared__ _Float16 sH[8][16 * HDIM];     // per-wave h tile (C->A relayout)
    __shared__ float    sVN[BGRAPHS * 12];    // block-local vn accumulator

    int tid = threadIdx.x;
    for (int i = tid; i < HDIM * HDIM; i += 256) sW1[i] = (_Float16)W1[i];
    for (int i = tid; i < HDIM * 16; i += 256) {
        int k = i >> 4, c = i & 15;
        sW2[i] = (c < 4) ? (_Float16)W2[k * 4 + c] : (_Float16)0.0f;
    }
    if (tid < HDIM) sB1[tid] = b1v[tid];
    if (tid < 4)    sB2[tid] = b2v[tid];
    for (int i = tid; i < BGRAPHS * 12; i += 256) sVN[i] = 0.0f;
    __syncthreads();

    int lane = tid & 31;
    int wave = tid >> 5;
    int hi   = lane >> 4;     // half-wave selector
    int lo   = lane & 15;

    // Register-resident B fragments (contiguous k order used here, same as A: lanes 0-15 K=0..15,
    // lanes 16-31 K=16..31, pairs (2v,2v+1) per VGPR).
    v16h bW1[4][2];
    v16h bW2[2];
    int bkhalf = hi * 16;
    for (int t = 0; t < 4; ++t)
        for (int s = 0; s < 2; ++s)
            for (int v = 0; v < 8; ++v) {
                int k = s * 32 + bkhalf + 2 * v;
                bW1[t][s][2 * v + 0] = sW1[(k + 0) * HDIM + t * 16 + lo];
                bW1[t][s][2 * v + 1] = sW1[(k + 1) * HDIM + t * 16 + lo];
            }
    for (int s = 0; s < 2; ++s)
        for (int v = 0; v < 8; ++v) {
            int k = s * 32 + bkhalf + 2 * v;
            bW2[s][2 * v + 0] = sW2[(k + 0) * 16 + lo];
            bW2[s][2 * v + 1] = sW2[(k + 1) * 16 + lo];
        }

    int blockStart = blockIdx.x * NODES_PER_BLOCK;
    _Float16* hbuf = sH[wave];
    int akgrp = hi * 16;      // A layout K half offset for this half-wave

    for (int it = 0; it < 8; ++it) {
        int tileBase = blockStart + (it * 8 + wave) * 16;   // wave-uniform
        if (tileBase >= n) break;

        // ---- GEMM1: h[16x64] = x[16x64] @ W1, f16 in / f32 acc ----
        int arow = min(tileBase + lo, n - 1);
        const float* xrow = feat + (size_t)arow * HDIM;
        v8f acc[4] = {};
        for (int s = 0; s < 2; ++s) {
            v16h afrag;
            for (int v = 0; v < 8; ++v) {
                int k0 = 2 * v + akgrp + s * 32;
                float2 xv = *(const float2*)(xrow + k0);   // 8B-aligned pair
                afrag[2 * v + 0] = (_Float16)xv.x;
                afrag[2 * v + 1] = (_Float16)xv.y;
            }
            for (int t = 0; t < 4; ++t)
                acc[t] = __builtin_amdgcn_wmma_f32_16x16x32_f16(
                    false, afrag, false, bW1[t][s], (short)0, acc[t], false, false);
        }

        // bias + SiLU, write back in A layout via per-wave LDS
        for (int t = 0; t < 4; ++t) {
            int ncol = t * 16 + lo;
            float bb = sB1[ncol];
            for (int j = 0; j < 8; ++j) {
                int m = hi * 8 + j;
                float hval = acc[t][j] + bb;
                float sg = 1.0f / (1.0f + __expf(-hval));
                hbuf[m * HDIM + ncol] = (_Float16)(hval * sg);
            }
        }
        asm volatile("s_wait_dscnt 0x0" ::: "memory");   // intra-wave LDS RAW

        // ---- GEMM2: w[16x16(pad)] = h @ W2 ----
        v8f accw = {};
        for (int s = 0; s < 2; ++s) {
            v16h afrag;
            for (int v = 0; v < 8; ++v) {
                int k0 = 2 * v + akgrp + s * 32;
                afrag[2 * v + 0] = hbuf[lo * HDIM + k0];
                afrag[2 * v + 1] = hbuf[lo * HDIM + k0 + 1];
            }
            accw = __builtin_amdgcn_wmma_f32_16x16x32_f16(
                false, afrag, false, bW2[s], (short)0, accw, false, false);
        }

        // ---- disp = (w + b2) * (pos - center), LDS segment accumulate ----
        if (lo < 4) {
            int c = lo;
            float bb2 = sB2[c];
            for (int j = 0; j < 8; ++j) {
                int node = tileBase + hi * 8 + j;
                if (node < n) {
                    int g = batch2[2 * node];
                    float wv = accw[j] + bb2;
                    float px = pos[node * 3 + 0] - center[g * 3 + 0];
                    float py = pos[node * 3 + 1] - center[g * 3 + 1];
                    float pz = pos[node * 3 + 2] - center[g * 3 + 2];
                    atomicAdd(&sVN[g * 12 + c * 3 + 0], wv * px);
                    atomicAdd(&sVN[g * 12 + c * 3 + 1], wv * py);
                    atomicAdd(&sVN[g * 12 + c * 3 + 2], wv * pz);
                }
            }
        }
    }

    __syncthreads();
    int last = min(blockStart + NODES_PER_BLOCK, n) - 1;
    int gmin = batch2[2 * blockStart];
    int gmax = batch2[2 * last];
    int span = (gmax - gmin + 1) * 12;
    for (int i = tid; i < span; i += 256) {
        float v = sVN[gmin * 12 + i];
        if (v != 0.0f) atomicAdd(&vn_sum[gmin * 12 + i], v);
    }
}

// --------------------------------------------------------------- finalize
__global__ __launch_bounds__(256) void vn_final(
    const float* __restrict__ vn_sum, const float* __restrict__ cs,
    const float* __restrict__ center, float* __restrict__ out) {
    int idx = blockIdx.x * 256 + threadIdx.x;   // b*4 + c
    if (idx < BGRAPHS * 4) {
        int b = idx >> 2, c = idx & 3;
        float cnt = fmaxf(cs[b * 4 + 3], 1.0f);
        float vx = vn_sum[b * 12 + c * 3 + 0] / cnt;
        float vy = vn_sum[b * 12 + c * 3 + 1] / cnt;
        float vz = vn_sum[b * 12 + c * 3 + 2] / cnt;
        float nrm = fmaxf(sqrtf(vx * vx + vy * vy + vz * vz), EPSF);
        out[b * 12 + c * 3 + 0] = vx / nrm + center[b * 3 + 0];
        out[b * 12 + c * 3 + 1] = vy / nrm + center[b * 3 + 1];
        out[b * 12 + c * 3 + 2] = vz / nrm + center[b * 3 + 2];
    }
}

extern "C" void kernel_launch(void* const* d_in, const int* in_sizes, int n_in,
                              void* d_out, int out_size, void* d_ws, size_t ws_size,
                              hipStream_t stream) {
    const float* feat   = (const float*)d_in[0];
    const float* pos    = (const float*)d_in[1];
    const int*   batch2 = (const int*)d_in[2];   // int64 viewed as dword pairs
    const float* W1     = (const float*)d_in[3];
    const float* b1v    = (const float*)d_in[4];
    const float* W2     = (const float*)d_in[5];
    const float* b2v    = (const float*)d_in[6];
    float* out = (float*)d_out;
    int n = in_sizes[1] / 3;                      // node count from node_pos

    float* ws         = (float*)d_ws;
    float* center_sum = ws;                       // B*4  (xyz sum + count)
    float* center     = ws + BGRAPHS * 4;         // B*3
    float* vn_sum     = ws + BGRAPHS * 4 + BGRAPHS * 3;  // B*12
    int ws_floats = BGRAPHS * (4 + 3 + 12);       // 19456

    int nb = (n + NODES_PER_BLOCK - 1) / NODES_PER_BLOCK;

    vn_zero_kernel<<<(ws_floats + 255) / 256, 256, 0, stream>>>(ws, ws_floats);
    vn_center_partial<<<nb, 256, 0, stream>>>(pos, batch2, center_sum, n);
    vn_center_final<<<(BGRAPHS + 255) / 256, 256, 0, stream>>>(center_sum, center);
    vn_main<<<nb, 256, 0, stream>>>(feat, pos, batch2, W1, b1v, W2, b2v,
                                    center, vn_sum, n);
    vn_final<<<(BGRAPHS * 4 + 255) / 256, 256, 0, stream>>>(vn_sum, center_sum, center, out);
}
